// UNet_config3_withMish_21646635171984
// MI455X (gfx1250) — matrix operand order repaired and checked by reading, not compiled, not run on hardware
//
#include <hip/hip_runtime.h>
#include <hip/hip_bf16.h>

// ---------------------------------------------------------------------------
// UNet forward (B=4, 256x256) for gfx1250 (MI455X).
// All heavy convs run as implicit-GEMM on v_wmma_f32_16x16x32_bf16 (wave32).
// Activations/weights: bf16 NHWC; accumulation: f32.
// ---------------------------------------------------------------------------

typedef __bf16 bf16;
typedef bf16  bf16x8  __attribute__((ext_vector_type(8)));
typedef bf16  bf16x16 __attribute__((ext_vector_type(16)));
typedef float f32x8   __attribute__((ext_vector_type(8)));
typedef int   i32x8   __attribute__((ext_vector_type(8)));

union AFrag { bf16x16 v; bf16x8 h[2]; };
union BZero { bf16x16 v; i32x8  i; };

__device__ __forceinline__ float mishf(float v) {
  float sp = (v > 20.0f) ? v : log1pf(expf(v));
  return v * tanhf(sp);
}

// ---------------------------------------------------------------------------
// Weight repack: OIHW f32 -> [Co][ky*kx][Ci] bf16 (channel-contiguous for WMMA)
// ---------------------------------------------------------------------------
__global__ void repack_w_kernel(const float* __restrict__ w, bf16* __restrict__ wp,
                                int Co, int Ci, int KH, int KW) {
  int t = blockIdx.x * 256 + threadIdx.x;
  int total = Co * Ci * KH * KW;
  if (t >= total) return;
  int kx = t % KW;
  int ky = (t / KW) % KH;
  int ci = (t / (KW * KH)) % Ci;
  int co = t / (KW * KH * Ci);
  wp[((co * KH * KW) + (ky * KW + kx)) * Ci + ci] = (bf16)w[t];
}

// ---------------------------------------------------------------------------
// Implicit-GEMM conv, KHxKW, stride 1, arbitrary pad. NHWC bf16 -> NHWC bf16.
// GEMM: M = Cout, N = B*Hout*Wout, K = Cin*KH*KW (Cin % 32 == 0).
// Block: 256 threads = 8 waves; block tile = 32(M) x 256(N); wave = 16 x 64.
// Interior fast path (no predication) + global_prefetch for next tap.
// Fused bias + mish epilogue.
// ---------------------------------------------------------------------------
__global__ __launch_bounds__(256) void conv_wmma_kernel(
    const bf16* __restrict__ in, const bf16* __restrict__ wp,
    const float* __restrict__ bias, bf16* __restrict__ out,
    int B, int Hin, int Win, int Hout, int Wout,
    int Cin, int Cout, int KH, int KW, int pad) {
  const int lane  = threadIdx.x & 31;
  const int wave  = threadIdx.x >> 5;
  const int wm    = wave & 1;                 // M half within block tile
  const int wn    = wave >> 1;                // N quarter (64 pixels)
  const int co0   = blockIdx.y * 32 + wm * 16;
  const int nbase = blockIdx.x * 256 + wn * 64;
  const int col   = lane & 15;
  const int khalf = lane >> 4;
  const int Np    = Hout * Wout;

  // decode the pixel owned by this lane in each of the 4 sub-tiles
  int bb[4], iy0[4], ix0[4];
#pragma unroll
  for (int s = 0; s < 4; ++s) {
    int n  = nbase + s * 16 + col;
    int b  = n / Np;
    int r  = n - b * Np;
    int oy = r / Wout;
    int ox = r - oy * Wout;
    bb[s] = b; iy0[s] = oy - pad; ix0[s] = ox - pad;
  }

  f32x8 zero8 = {0.f,0.f,0.f,0.f,0.f,0.f,0.f,0.f};
  f32x8 acc0 = zero8, acc1 = zero8, acc2 = zero8, acc3 = zero8;

  BZero zb; i32x8 zi = {0,0,0,0,0,0,0,0}; zb.i = zi;

  const int koffA = khalf * 16;   // A-frag K offset: contiguous k = 16*khalf + s (matches B)
  const int kbB   = khalf * 16;   // B-frag K offset: contiguous k = 16*khalf + s (matches A)
  const bf16* wrow = wp + (size_t)(co0 + col) * (size_t)(KH * KW) * (size_t)Cin;
  const int KK = KH * KW;

  for (int kyx = 0; kyx < KK; ++kyx) {
    const int ky = kyx / KW;
    const int kx = kyx - ky * KW;
    int  sboff[4];
    bool sval[4];
#pragma unroll
    for (int s = 0; s < 4; ++s) {
      int iy = iy0[s] + ky, ix = ix0[s] + kx;
      sval[s]  = (iy >= 0) && (iy < Hin) && (ix >= 0) && (ix < Win);
      sboff[s] = ((bb[s] * Hin + iy) * Win + ix) * Cin;
    }
    // Prefetch next tap's activation rows into cache (global_prefetch_b8).
    if (kyx + 1 < KK) {
      const int ky2 = (kyx + 1) / KW;
      const int kx2 = (kyx + 1) - ky2 * KW;
#pragma unroll
      for (int s = 0; s < 4; ++s) {
        int iy = iy0[s] + ky2, ix = ix0[s] + kx2;
        if (iy >= 0 && iy < Hin && ix >= 0 && ix < Win)
          __builtin_prefetch(in + ((bb[s] * Hin + iy) * Win + ix) * Cin + kbB, 0, 1);
      }
    }
    const bf16* wk = wrow + kyx * Cin;
    const bool allv = sval[0] & sval[1] & sval[2] & sval[3];
    if (allv) {
      // Interior fast path: unconditional, clause-friendly loads.
      for (int ci0 = 0; ci0 < Cin; ci0 += 32) {
        AFrag a;
        a.h[0] = *(const bf16x8*)(wk + ci0 + koffA);
        a.h[1] = *(const bf16x8*)(wk + ci0 + 8 + koffA);
        bf16x16 b0 = *(const bf16x16*)(in + sboff[0] + ci0 + kbB);
        bf16x16 b1 = *(const bf16x16*)(in + sboff[1] + ci0 + kbB);
        bf16x16 b2 = *(const bf16x16*)(in + sboff[2] + ci0 + kbB);
        bf16x16 b3 = *(const bf16x16*)(in + sboff[3] + ci0 + kbB);
        acc0 = __builtin_amdgcn_wmma_f32_16x16x32_bf16(false, a.v, false, b0, (short)0, acc0, false, false);
        acc1 = __builtin_amdgcn_wmma_f32_16x16x32_bf16(false, a.v, false, b1, (short)0, acc1, false, false);
        acc2 = __builtin_amdgcn_wmma_f32_16x16x32_bf16(false, a.v, false, b2, (short)0, acc2, false, false);
        acc3 = __builtin_amdgcn_wmma_f32_16x16x32_bf16(false, a.v, false, b3, (short)0, acc3, false, false);
      }
    } else {
      // Halo path: per-sub-tile predicated loads with zero fill.
      for (int ci0 = 0; ci0 < Cin; ci0 += 32) {
        AFrag a;
        a.h[0] = *(const bf16x8*)(wk + ci0 + koffA);
        a.h[1] = *(const bf16x8*)(wk + ci0 + 8 + koffA);
        bf16x16 b0 = zb.v, b1 = zb.v, b2 = zb.v, b3 = zb.v;
        if (sval[0]) b0 = *(const bf16x16*)(in + sboff[0] + ci0 + kbB);
        if (sval[1]) b1 = *(const bf16x16*)(in + sboff[1] + ci0 + kbB);
        if (sval[2]) b2 = *(const bf16x16*)(in + sboff[2] + ci0 + kbB);
        if (sval[3]) b3 = *(const bf16x16*)(in + sboff[3] + ci0 + kbB);
        acc0 = __builtin_amdgcn_wmma_f32_16x16x32_bf16(false, a.v, false, b0, (short)0, acc0, false, false);
        acc1 = __builtin_amdgcn_wmma_f32_16x16x32_bf16(false, a.v, false, b1, (short)0, acc1, false, false);
        acc2 = __builtin_amdgcn_wmma_f32_16x16x32_bf16(false, a.v, false, b2, (short)0, acc2, false, false);
        acc3 = __builtin_amdgcn_wmma_f32_16x16x32_bf16(false, a.v, false, b3, (short)0, acc3, false, false);
      }
    }
  }

  // Epilogue: bias + mish, write 8 contiguous channels (16B) per lane/sub-tile.
  float bv[8];
#pragma unroll
  for (int i = 0; i < 8; ++i) bv[i] = bias[co0 + khalf * 8 + i];

  f32x8 accs[4] = {acc0, acc1, acc2, acc3};
#pragma unroll
  for (int s = 0; s < 4; ++s) {
    int n = nbase + s * 16 + col;
    bf16x8 ov;
#pragma unroll
    for (int i = 0; i < 8; ++i) ov[i] = (bf16)mishf(accs[s][i] + bv[i]);
    *(bf16x8*)(out + (size_t)n * Cout + co0 + khalf * 8) = ov;
  }
}

// ---------------------------------------------------------------------------
// Per-sample dynamic conv (Cin=1) + mish. x: NCHW f32, w: (B,Co,K,K) f32.
// out: NHWC bf16 (Hout = Hin + 2*pad - K + 1).
// ---------------------------------------------------------------------------
__global__ void dyn_conv_mish_kernel(const float* __restrict__ x, const float* __restrict__ w,
                                     bf16* __restrict__ out,
                                     int B, int Hin, int Win, int Hout, int Wout,
                                     int Co, int K, int pad) {
  int t = blockIdx.x * 256 + threadIdx.x;
  int total = B * Hout * Wout * Co;
  if (t >= total) return;
  int co  = t % Co;
  int pix = t / Co;
  int ox  = pix % Wout;
  int tmp = pix / Wout;
  int oy  = tmp % Hout;
  int b   = tmp / Hout;
  float acc = 0.f;
  for (int ky = 0; ky < K; ++ky) {
    int iy = oy + ky - pad;
    if (iy < 0 || iy >= Hin) continue;
    for (int kx = 0; kx < K; ++kx) {
      int ix = ox + kx - pad;
      if (ix < 0 || ix >= Win) continue;
      acc += x[(b * Hin + iy) * Win + ix] * w[((b * Co + co) * K + ky) * K + kx];
    }
  }
  out[t] = (bf16)mishf(acc);
}

// ---------------------------------------------------------------------------
// 2x2 maxpool, NHWC bf16.
// ---------------------------------------------------------------------------
__global__ void maxpool_kernel(const bf16* __restrict__ in, bf16* __restrict__ out,
                               int B, int Hi, int Wi, int C) {
  int Ho = Hi >> 1, Wo = Wi >> 1;
  int total = B * Ho * Wo * C;
  int t = blockIdx.x * 256 + threadIdx.x;
  if (t >= total) return;
  int c   = t % C;
  int pix = t / C;
  int ox  = pix % Wo;
  int tmp = pix / Wo;
  int oy  = tmp % Ho;
  int b   = tmp / Ho;
  const bf16* p = in + ((size_t)(b * Hi + 2 * oy) * Wi + 2 * ox) * C + c;
  float m = (float)p[0];
  m = fmaxf(m, (float)p[C]);
  m = fmaxf(m, (float)p[(size_t)Wi * C]);
  m = fmaxf(m, (float)p[(size_t)Wi * C + C]);
  out[t] = (bf16)m;
}

// ---------------------------------------------------------------------------
// Batchnorm (batch stats): one block per channel computes mean / rsqrt(var+eps).
// ---------------------------------------------------------------------------
__global__ void bn_stats_kernel(const bf16* __restrict__ h, float* __restrict__ mean,
                                float* __restrict__ rstd, int NPix, int C) {
  __shared__ float ss[256], sq[256];
  int c = blockIdx.x;
  float s = 0.f, q = 0.f;
  for (int i = threadIdx.x; i < NPix; i += 256) {
    float v = (float)h[(size_t)i * C + c];
    s += v; q += v * v;
  }
  ss[threadIdx.x] = s; sq[threadIdx.x] = q;
  __syncthreads();
  for (int st = 128; st > 0; st >>= 1) {
    if ((int)threadIdx.x < st) {
      ss[threadIdx.x] += ss[threadIdx.x + st];
      sq[threadIdx.x] += sq[threadIdx.x + st];
    }
    __syncthreads();
  }
  if (threadIdx.x == 0) {
    float mu  = ss[0] / (float)NPix;
    float var = sq[0] / (float)NPix - mu * mu;
    mean[c] = mu;
    rstd[c] = rsqrtf(var + 1e-5f);
  }
}

__global__ void bn_apply_kernel(bf16* __restrict__ h, const float* __restrict__ g,
                                const float* __restrict__ be, const float* __restrict__ mean,
                                const float* __restrict__ rstd, int total, int C) {
  int t = blockIdx.x * 256 + threadIdx.x;
  if (t >= total) return;
  int c = t % C;
  float v = (float)h[t];
  h[t] = (bf16)(g[c] * (v - mean[c]) * rstd[c] + be[c]);
}

// ---------------------------------------------------------------------------
// Bilinear x2 upsample (align_corners=True) of `low` fused with channel-concat
// of `skip`: out[..., 0:C1] = up2(low), out[..., C1:C1+C2] = skip. NHWC bf16.
// ---------------------------------------------------------------------------
__global__ void up_concat_kernel(const bf16* __restrict__ low, const bf16* __restrict__ skip,
                                 bf16* __restrict__ out, int B, int H, int W, int C1, int C2) {
  int Ho = 2 * H, Wo = 2 * W, Ct = C1 + C2;
  int total = B * Ho * Wo * Ct;
  int t = blockIdx.x * 256 + threadIdx.x;
  if (t >= total) return;
  int c   = t % Ct;
  int pix = t / Ct;
  int ox  = pix % Wo;
  int tmp = pix / Wo;
  int oy  = tmp % Ho;
  int b   = tmp / Ho;
  float r;
  if (c < C1) {
    float sc = (float)(H - 1) / (float)(Ho - 1);   // align_corners (H==W here)
    float sy = oy * sc, sx = ox * sc;
    int y0 = (int)sy, x0 = (int)sx;
    int y1 = (y0 + 1 < H) ? y0 + 1 : H - 1;
    int x1 = (x0 + 1 < W) ? x0 + 1 : W - 1;
    float fy = sy - (float)y0, fx = sx - (float)x0;
    const bf16* base = low + (size_t)(b * H) * W * C1;
    float v00 = (float)base[((size_t)y0 * W + x0) * C1 + c];
    float v10 = (float)base[((size_t)y1 * W + x0) * C1 + c];
    float v01 = (float)base[((size_t)y0 * W + x1) * C1 + c];
    float v11 = (float)base[((size_t)y1 * W + x1) * C1 + c];
    float a = v00 * (1.f - fy) + v10 * fy;
    float d = v01 * (1.f - fy) + v11 * fy;
    r = a * (1.f - fx) + d * fx;
  } else {
    r = (float)skip[((size_t)(b * Ho + oy) * Wo + ox) * C2 + (c - C1)];
  }
  out[t] = (bf16)r;
}

// ---------------------------------------------------------------------------
// Final 1x1 conv, Cout=1: NHWC bf16 -> f32 (pixel-major == NCHW for C=1).
// ---------------------------------------------------------------------------
__global__ void conv1x1_out_kernel(const bf16* __restrict__ in, const float* __restrict__ wl,
                                   const float* __restrict__ bl, float* __restrict__ out,
                                   int total, int C) {
  int t = blockIdx.x * 256 + threadIdx.x;
  if (t >= total) return;
  const bf16* p = in + (size_t)t * C;
  float acc = bl[0];
  for (int c = 0; c < C; ++c) acc += (float)p[c] * wl[c];
  out[t] = acc;
}

// ---------------------------------------------------------------------------
// Host orchestration
// ---------------------------------------------------------------------------
extern "C" void kernel_launch(void* const* d_in, const int* in_sizes, int n_in,
                              void* d_out, int out_size, void* d_ws, size_t ws_size,
                              hipStream_t stream) {
  (void)in_sizes; (void)n_in; (void)out_size; (void)ws_size;
  const int Bn = 4;

  const float* x    = (const float*)d_in[0];
  const float* wdyn = (const float*)d_in[1];
  const float* d1w  = (const float*)d_in[2];
  const float* d1b  = (const float*)d_in[3];
  const float* w2a  = (const float*)d_in[4];
  const float* b2a  = (const float*)d_in[5];
  const float* w2b  = (const float*)d_in[6];
  const float* b2b  = (const float*)d_in[7];
  const float* w3a  = (const float*)d_in[8];
  const float* b3a  = (const float*)d_in[9];
  const float* w3b  = (const float*)d_in[10];
  const float* b3b  = (const float*)d_in[11];
  const float* w4a  = (const float*)d_in[12];
  const float* b4a  = (const float*)d_in[13];
  const float* w4b  = (const float*)d_in[14];
  const float* b4b  = (const float*)d_in[15];
  const float* g4   = (const float*)d_in[16];
  const float* be4  = (const float*)d_in[17];
  const float* u3a  = (const float*)d_in[18];
  const float* ub3a = (const float*)d_in[19];
  const float* u3b  = (const float*)d_in[20];
  const float* ub3b = (const float*)d_in[21];
  const float* u2a  = (const float*)d_in[22];
  const float* ub2a = (const float*)d_in[23];
  const float* u2b  = (const float*)d_in[24];
  const float* ub2b = (const float*)d_in[25];
  const float* u1a  = (const float*)d_in[26];
  const float* ub1a = (const float*)d_in[27];
  const float* u1b  = (const float*)d_in[28];
  const float* ub1b = (const float*)d_in[29];
  const float* wl   = (const float*)d_in[30];
  const float* bl   = (const float*)d_in[31];

  // ---- workspace layout: bump-allocate persistent buffers, overlap scratch ----
  char* wsb = (char*)d_ws;
  size_t off = 0;
  auto takeB  = [&](size_t bytes) -> char* {
    char* p = wsb + off;
    off = (off + bytes + 255) & ~(size_t)255;
    return p;
  };
  auto takeBF = [&](size_t elems) -> bf16* { return (bf16*)takeB(elems * 2); };

  // packed weights (persistent)
  bf16* wp_d1  = takeBF((size_t)32  * 16 * 32);
  bf16* wp_2a  = takeBF((size_t)128 * 9  * 32);
  bf16* wp_2b  = takeBF((size_t)128 * 9  * 128);
  bf16* wp_3a  = takeBF((size_t)256 * 9  * 128);
  bf16* wp_3b  = takeBF((size_t)256 * 9  * 256);
  bf16* wp_4a  = takeBF((size_t)512 * 9  * 256);
  bf16* wp_4b  = takeBF((size_t)512 * 9  * 512);
  bf16* wp_u3a = takeBF((size_t)256 * 9  * 768);
  bf16* wp_u3b = takeBF((size_t)256 * 9  * 256);
  bf16* wp_u2a = takeBF((size_t)128 * 9  * 384);
  bf16* wp_u2b = takeBF((size_t)128 * 9  * 128);
  bf16* wp_u1a = takeBF((size_t)32  * 9  * 160);
  bf16* wp_u1b = takeBF((size_t)32  * 9  * 32);

  // skip tensors (persistent until consumed by concat)
  bf16* c1 = takeBF((size_t)Bn * 256 * 256 * 32);
  bf16* c2 = takeBF((size_t)Bn * 128 * 128 * 128);
  bf16* c3 = takeBF((size_t)Bn * 64  * 64  * 256);

  // overlapped scratch region
  const size_t MB = (size_t)1 << 20;
  const size_t S  = off;
  bf16*  act0   = (bf16*)(wsb + S);              // 16.7 MB
  bf16*  p1     = (bf16*)(wsb + S);              // 4.2 MB (act0 dead)
  bf16*  t2a    = (bf16*)(wsb + S + 5  * MB);    // 16.8 MB
  bf16*  p2     = (bf16*)(wsb + S);              // 4.2 MB
  bf16*  t3a    = (bf16*)(wsb + S + 5  * MB);    // 8.4 MB
  bf16*  p3     = (bf16*)(wsb + S);              // 2.1 MB
  bf16*  t4a    = (bf16*)(wsb + S + 3  * MB);    // 4.2 MB
  bf16*  h4     = (bf16*)(wsb + S + 8  * MB);    // 4.2 MB
  float* bnmean = (float*)(wsb + S + 13 * MB);   // 2 KB
  float* bnrstd = bnmean + 512;
  bf16*  cat3   = (bf16*)(wsb + S + 14 * MB);    // 25.2 MB
  bf16*  tu3a   = (bf16*)(wsb + S);              // 8.4 MB
  bf16*  u3o    = (bf16*)(wsb + S + 9  * MB);    // 8.4 MB
  bf16*  cat2   = (bf16*)(wsb + S + 18 * MB);    // 50.4 MB
  bf16*  tu2a   = (bf16*)(wsb + S);              // 16.8 MB
  bf16*  u2o    = (bf16*)(wsb + S + 18 * MB);    // 16.8 MB (cat2 dead)
  bf16*  cat1   = (bf16*)(wsb + S + 36 * MB);    // 83.9 MB
  bf16*  tu1a   = (bf16*)(wsb + S);              // 16.8 MB
  bf16*  u1o    = (bf16*)(wsb + S + 18 * MB);    // 16.8 MB

  auto cdiv = [](int a, int b) { return (a + b - 1) / b; };

  auto rep = [&](const float* w, bf16* wp, int Co, int Ci, int KH, int KW) {
    int tot = Co * Ci * KH * KW;
    repack_w_kernel<<<cdiv(tot, 256), 256, 0, stream>>>(w, wp, Co, Ci, KH, KW);
  };
  auto conv = [&](const bf16* in, const bf16* wp, const float* bias, bf16* out,
                  int Hin, int Win, int Hout, int Wout, int Cin, int Cout, int K, int pad) {
    int N = Bn * Hout * Wout;              // multiple of 256 for all layers here
    dim3 g(N / 256, Cout / 32);
    conv_wmma_kernel<<<g, 256, 0, stream>>>(in, wp, bias, out,
                                            Bn, Hin, Win, Hout, Wout, Cin, Cout, K, K, pad);
  };
  auto pool = [&](const bf16* in, bf16* out, int Hi, int C) {
    int tot = Bn * (Hi / 2) * (Hi / 2) * C;
    maxpool_kernel<<<cdiv(tot, 256), 256, 0, stream>>>(in, out, Bn, Hi, Hi, C);
  };
  auto upcat = [&](const bf16* low, const bf16* skip, bf16* out, int H, int C1, int C2) {
    int tot = Bn * (2 * H) * (2 * H) * (C1 + C2);
    up_concat_kernel<<<cdiv(tot, 256), 256, 0, stream>>>(low, skip, out, Bn, H, H, C1, C2);
  };

  // ---- weight repack (every call; deterministic & graph-safe) ----
  rep(d1w, wp_d1, 32, 32, 4, 4);
  rep(w2a, wp_2a, 128, 32, 3, 3);   rep(w2b, wp_2b, 128, 128, 3, 3);
  rep(w3a, wp_3a, 256, 128, 3, 3);  rep(w3b, wp_3b, 256, 256, 3, 3);
  rep(w4a, wp_4a, 512, 256, 3, 3);  rep(w4b, wp_4b, 512, 512, 3, 3);
  rep(u3a, wp_u3a, 256, 768, 3, 3); rep(u3b, wp_u3b, 256, 256, 3, 3);
  rep(u2a, wp_u2a, 128, 384, 3, 3); rep(u2b, wp_u2b, 128, 128, 3, 3);
  rep(u1a, wp_u1a, 32, 160, 3, 3);  rep(u1b, wp_u1b, 32, 32, 3, 3);

  // ---- encoder ----
  {
    int tot = Bn * 255 * 255 * 32;   // dyn_conv: pad = k/2-1 = 1 -> 255x255
    dyn_conv_mish_kernel<<<cdiv(tot, 256), 256, 0, stream>>>(
        x, wdyn, act0, Bn, 256, 256, 255, 255, 32, 4, 1);
  }
  conv(act0, wp_d1, d1b, c1, 255, 255, 256, 256, 32, 32, 4, 2);     // c1 (B,256,256,32)
  pool(c1, p1, 256, 32);
  conv(p1,  wp_2a, b2a, t2a, 128, 128, 128, 128, 32, 128, 3, 1);
  conv(t2a, wp_2b, b2b, c2,  128, 128, 128, 128, 128, 128, 3, 1);   // c2
  pool(c2, p2, 128, 128);
  conv(p2,  wp_3a, b3a, t3a, 64, 64, 64, 64, 128, 256, 3, 1);
  conv(t3a, wp_3b, b3b, c3,  64, 64, 64, 64, 256, 256, 3, 1);       // c3
  pool(c3, p3, 64, 256);
  conv(p3,  wp_4a, b4a, t4a, 32, 32, 32, 32, 256, 512, 3, 1);
  conv(t4a, wp_4b, b4b, h4,  32, 32, 32, 32, 512, 512, 3, 1);       // h (B,32,32,512)

  // ---- batchnorm (batch statistics) ----
  bn_stats_kernel<<<512, 256, 0, stream>>>(h4, bnmean, bnrstd, Bn * 32 * 32, 512);
  {
    int tot = Bn * 32 * 32 * 512;
    bn_apply_kernel<<<cdiv(tot, 256), 256, 0, stream>>>(h4, g4, be4, bnmean, bnrstd, tot, 512);
  }

  // ---- decoder ----
  upcat(h4, c3, cat3, 32, 512, 256);                                 // (B,64,64,768)
  conv(cat3, wp_u3a, ub3a, tu3a, 64, 64, 64, 64, 768, 256, 3, 1);
  conv(tu3a, wp_u3b, ub3b, u3o,  64, 64, 64, 64, 256, 256, 3, 1);

  upcat(u3o, c2, cat2, 64, 256, 128);                                // (B,128,128,384)
  conv(cat2, wp_u2a, ub2a, tu2a, 128, 128, 128, 128, 384, 128, 3, 1);
  conv(tu2a, wp_u2b, ub2b, u2o,  128, 128, 128, 128, 128, 128, 3, 1);

  upcat(u2o, c1, cat1, 128, 128, 32);                                // (B,256,256,160)
  conv(cat1, wp_u1a, ub1a, tu1a, 256, 256, 256, 256, 160, 32, 3, 1);
  conv(tu1a, wp_u1b, ub1b, u1o,  256, 256, 256, 256, 32, 32, 3, 1);

  // ---- final 1x1 conv -> f32 output ----
  {
    int tot = Bn * 256 * 256;
    conv1x1_out_kernel<<<cdiv(tot, 256), 256, 0, stream>>>(u1o, wl, bl, (float*)d_out, tot, 32);
  }
}
